// GcnDenseModel_41927470743865
// MI455X (gfx1250) — compile-verified, matrix operand order checked by reading, not run on hardware
//
#include <hip/hip_runtime.h>
#include <hip/hip_bf16.h>

// ---------------- problem constants (match reference) ----------------
#define NN     8192      // nodes == seq len
#define EE     131072    // edges (before self loops)
#define FIN    512       // input feature size
#define FOUT   256       // GCN out / LSTM input
#define EDIM   128       // edge attr dim
#define HID    256       // LSTM hidden
#define FOURH  1024      // 4*HID
#define NPOOL  (NN * (HID / 2))          // 1,048,576
#define LININ  (2 * NPOOL + 3 * HID)     // 2,097,920

// ---------------- CDNA5 WMMA types ----------------
typedef __attribute__((ext_vector_type(16))) __bf16 v16bf;
typedef __attribute__((ext_vector_type(8)))  __bf16 v8bf;
typedef __attribute__((ext_vector_type(8)))  float  v8f;

__device__ __forceinline__ unsigned short f2bf(float f) {
  unsigned int u = __float_as_uint(f);
  unsigned int r = u + 0x7FFFu + ((u >> 16) & 1u);   // round to nearest even
  return (unsigned short)(r >> 16);
}

// Load one lane's 16-bf16 A/B fragment (16x32 tile, rows of length `ld`).
// Per ISA: lanes 0-15 hold K=[0..7],[16..23]; lanes 16-31 hold K=[8..15],[24..31].
__device__ __forceinline__ v16bf load_frag(const unsigned short* rowp, int kbase) {
  union { v16bf v; v8bf h[2]; } u;
  u.h[0] = *(const v8bf*)(rowp + kbase);
  u.h[1] = *(const v8bf*)(rowp + kbase + 16);
  return u.v;
}

__device__ __forceinline__ v8f v8f_zero() {
  v8f a;
#pragma unroll
  for (int i = 0; i < 8; ++i) a[i] = 0.0f;
  return a;
}

__device__ __forceinline__ float sigmoidf_(float x) { return 1.0f / (1.0f + __expf(-x)); }

// ---------------- utility kernels ----------------
__global__ void k_f32_to_bf16(const float* __restrict__ src,
                              unsigned short* __restrict__ dst, long n) {
  long i = (long)blockIdx.x * blockDim.x + threadIdx.x;
  long stride = (long)gridDim.x * blockDim.x;
  for (; i < n; i += stride) dst[i] = f2bf(src[i]);
}

__global__ void k_deg_init(float* deg, int n) {
  int i = blockIdx.x * blockDim.x + threadIdx.x;
  if (i < n) deg[i] = 1.0f;   // self loop
}

__global__ void k_deg_accum(const int* __restrict__ col, float* deg, int e) {
  int i = blockIdx.x * blockDim.x + threadIdx.x;
  if (i < e) atomicAdd(&deg[col[i]], 1.0f);
}

__global__ void k_dis(const float* __restrict__ deg, float* dis, int n) {
  int i = blockIdx.x * blockDim.x + threadIdx.x;
  if (i < n) dis[i] = rsqrtf(deg[i]);   // deg >= 1 always
}

__global__ void k_w2_rowsum(const float* __restrict__ W2, float* __restrict__ w2sum) {
  int j = blockIdx.x * blockDim.x + threadIdx.x;
  if (j < FOUT) {
    float s = 0.0f;
    const float* r = W2 + (size_t)j * EDIM;
#pragma unroll 8
    for (int k = 0; k < EDIM; ++k) s += r[k];
    w2sum[j] = s;
  }
}

// ---------------- WMMA GEMM: C[M,N] = A[M,K] @ B[N,K]^T (bf16 in, f32 out) ----
// Block = 256 threads (8 waves). One block computes a 128(M) x 16(N) strip.
// The 16xK B (weight) tile is staged ONCE into LDS with async-to-LDS copies
// (ASYNCcnt-tracked, no VGPR staging) and shared by all 8 waves; the padded
// row stride (520 bf16 = 1040 B => 4-dword bank stride) makes the 16-lane
// b128 fragment reads from LDS bank-conflict-free. A rows stream from global
// (L2) with a speculative prefetch ahead of the consume point.
// grid = (N/16, M/128). Requires K % 32 == 0, M % 128 == 0, K <= 512.
#define BPAD 520
__global__ void __launch_bounds__(256)
k_wmma_gemm_bf16(const unsigned short* __restrict__ A,
                 const unsigned short* __restrict__ B,
                 float* __restrict__ C, int M, int N, int K) {
  __shared__ unsigned short bs[16][BPAD];
  const int t    = threadIdx.x;
  const int lane = t & 31;
  const int wave = t >> 5;
  const int l16  = lane & 15;
  const int half = lane >> 4;

  // ---- stage B tile [16 x K] -> LDS via GLOBAL_LOAD_ASYNC_TO_LDS_B128 ----
  {
    const int chunksPerRow = K >> 3;            // 8 bf16 per 16-byte chunk
    const int totalChunks  = chunksPerRow << 4; // 16 rows
    const unsigned short* Bbase = B + (size_t)blockIdx.x * 16 * K;
    for (int c = t; c < totalChunks; c += 256) {
      int row = c / chunksPerRow;
      int kc  = (c - row * chunksPerRow) << 3;
      unsigned lds = (unsigned)(uintptr_t)(&bs[row][kc]);       // wave-relative LDS offset
      unsigned long long ga =
          (unsigned long long)(uintptr_t)(Bbase + (size_t)row * K + kc);
      asm volatile("global_load_async_to_lds_b128 %0, %1, off"
                   :: "v"(lds), "v"(ga) : "memory");
    }
    asm volatile("s_wait_asynccnt 0x0" ::: "memory");
  }
  __syncthreads();

  const unsigned short* arow = A + (size_t)(blockIdx.y * 128 + wave * 16 + l16) * K;
  const unsigned short* brow = &bs[l16][0];
  v8f acc = v8f_zero();
  for (int k = 0; k < K; k += 32) {
    __builtin_prefetch(arow + k + 128, 0, 0);    // -> global_prefetch_b8 (speculative)
    v16bf a = load_frag(arow, k + half * 8);     // global stream
    v16bf b = load_frag(brow, k + half * 8);     // LDS (ds_load_b128)
    acc = __builtin_amdgcn_wmma_f32_16x16x32_bf16(false, a, false, b,
                                                  (short)0, acc, false, false);
  }
  int n = blockIdx.x * 16 + l16;
  size_t base = (size_t)(blockIdx.y * 128 + wave * 16 + half * 8) * N + n;
#pragma unroll
  for (int r = 0; r < 8; ++r) C[base + (size_t)r * N] = acc[r];
}

// ------- fused edge kernel: ew = ea@W2^T (WMMA) -> tanh/norm -> scatter -------
// Block = 256 threads (8 waves). Block handles 16 edges; wave w covers N-tiles
// {2w, 2w+1} so 8 waves cover all 256 output features.
__global__ void __launch_bounds__(256)
k_gcn_edges(const unsigned short* __restrict__ eaB,
            const unsigned short* __restrict__ w2B,
            const int* __restrict__ row, const int* __restrict__ col,
            const float* __restrict__ dis, const float* __restrict__ xw,
            float* __restrict__ agg) {
  int lane = threadIdx.x & 31;
  int wave = threadIdx.x >> 5;
  int l16  = lane & 15;
  int half = lane >> 4;
  int eBase = blockIdx.x * 16;
  const unsigned short* arow = eaB + (size_t)(eBase + l16) * EDIM;
#pragma unroll
  for (int s = 0; s < 2; ++s) {
    int nt = wave * 2 + s;
    const unsigned short* brow = w2B + (size_t)(nt * 16 + l16) * EDIM;
    v8f acc = v8f_zero();
#pragma unroll
    for (int k = 0; k < EDIM; k += 32) {
      v16bf a = load_frag(arow, k + half * 8);
      v16bf b = load_frag(brow, k + half * 8);
      acc = __builtin_amdgcn_wmma_f32_16x16x32_bf16(false, a, false, b,
                                                    (short)0, acc, false, false);
    }
    int n = nt * 16 + l16;
#pragma unroll
    for (int r = 0; r < 8; ++r) {
      int e  = eBase + half * 8 + r;
      int ro = row[e], co = col[e];
      float nrm = dis[ro] * dis[co];
      float m   = nrm * tanhf(xw[(size_t)ro * FOUT + n] * acc[r]);
      atomicAdd(&agg[(size_t)co * FOUT + n], m);
    }
  }
}

// ------- finalize: self-loop msg, mean, bias, sigmoid -> bf16 seq -------------
__global__ void k_gcn_finalize(const float* __restrict__ agg,
                               const float* __restrict__ xw,
                               const float* __restrict__ w2sum,
                               const float* __restrict__ deg,
                               const float* __restrict__ bias,
                               unsigned short* __restrict__ seqB) {
  long idx = (long)blockIdx.x * blockDim.x + threadIdx.x;
  long total = (long)NN * FOUT;
  long stride = (long)gridDim.x * blockDim.x;
  for (; idx < total; idx += stride) {
    int i = (int)(idx / FOUT), j = (int)(idx % FOUT);
    float dinv = 1.0f / deg[i];
    float sm = dinv * tanhf(xw[idx] * w2sum[j]);       // self-loop message
    float g  = (agg[idx] + sm) * dinv + bias[j];       // scatter-mean + bias
    seqB[idx] = f2bf(sigmoidf_(g));
  }
}

// ------- LSTM recurrence: persistent single workgroup, 1024 threads -----------
// Strict 8192-step dependency chain (M=1 matvec per step -> matrix unit can't
// help). Thread t owns gate row t of W_hh[1024,256]; W_hh streams from L2
// (1 MB/step, resident in 192 MB L2); h, c, gates live in LDS.
__global__ void __launch_bounds__(1024)
k_lstm_rec(const float* __restrict__ xp,        // [L, 1024] precomputed seq@W_ih^T
           const float* __restrict__ Whh,       // [1024, 256] fp32
           float* __restrict__ hs,              // [L, 256] fp32
           unsigned short* __restrict__ hsB,    // [L, 256] bf16 (may be null)
           int L) {
  __shared__ float h[HID];
  __shared__ float c[HID];
  __shared__ float gbuf[FOURH];
  int t = threadIdx.x;
  if (t < HID) { h[t] = 0.0f; c[t] = 0.0f; }
  __syncthreads();
  const float* wr = Whh + (size_t)t * HID;
  for (int step = 0; step < L; ++step) {
    float acc = xp[(size_t)step * FOURH + t];
#pragma unroll 8
    for (int k = 0; k < HID; ++k) acc += h[k] * wr[k];
    gbuf[t] = acc;
    __syncthreads();
    if (t < HID) {
      float ig = sigmoidf_(gbuf[t]);
      float fg = sigmoidf_(gbuf[HID + t]);
      float gg = tanhf(gbuf[2 * HID + t]);
      float og = sigmoidf_(gbuf[3 * HID + t]);
      float cc = fg * c[t] + ig * gg;
      float hh = og * tanhf(cc);
      c[t] = cc; h[t] = hh;
      hs[(size_t)step * HID + t] = hh;
      if (hsB) hsB[(size_t)step * HID + t] = f2bf(hh);
    }
    __syncthreads();
  }
}

// ------- pooled-feature dot with lin_W rows (3 x 2.1M), block-reduced ---------
__global__ void __launch_bounds__(256)
k_lin_reduce(const float* __restrict__ h0s, const float* __restrict__ h1s,
             const float* __restrict__ h2s, const float* __restrict__ linW,
             float* __restrict__ raw) {
  __shared__ float s0[256], s1[256], s2[256];
  float a0 = 0.f, a1 = 0.f, a2 = 0.f;
  long i = (long)blockIdx.x * blockDim.x + threadIdx.x;
  long stride = (long)gridDim.x * blockDim.x;
  for (; i < (long)LININ; i += stride) {
    float v;
    if (i < (long)NPOOL) {                     // adaptive max pool (pairs)
      long t = i / 128, q = i % 128;
      float a = h2s[t * HID + 2 * q], b = h2s[t * HID + 2 * q + 1];
      v = fmaxf(a, b);
    } else if (i < 2L * NPOOL) {               // adaptive avg pool
      long j = i - NPOOL;
      long t = j / 128, q = j % 128;
      v = 0.5f * (h2s[t * HID + 2 * q] + h2s[t * HID + 2 * q + 1]);
    } else {                                   // h0T, h1T, h2T
      long j = i - 2L * NPOOL;
      const long last = (long)(NN - 1) * HID;
      v = (j < HID) ? h0s[last + j]
        : (j < 2 * HID) ? h1s[last + (j - HID)]
        : h2s[last + (j - 2 * HID)];
    }
    a0 += linW[i] * v;
    a1 += linW[(long)LININ + i] * v;
    a2 += linW[2L * LININ + i] * v;
  }
  int t = threadIdx.x;
  s0[t] = a0; s1[t] = a1; s2[t] = a2;
  __syncthreads();
  for (int off = 128; off > 0; off >>= 1) {
    if (t < off) { s0[t] += s0[t + off]; s1[t] += s1[t + off]; s2[t] += s2[t + off]; }
    __syncthreads();
  }
  if (t == 0) {
    atomicAdd(&raw[0], s0[0]);
    atomicAdd(&raw[1], s1[0]);
    atomicAdd(&raw[2], s2[0]);
  }
}

__global__ void k_softmax3(const float* __restrict__ raw,
                           const float* __restrict__ lin_b,
                           float* __restrict__ out) {
  if (threadIdx.x == 0 && blockIdx.x == 0) {
    float l0 = raw[0] + lin_b[0], l1 = raw[1] + lin_b[1], l2 = raw[2] + lin_b[2];
    float m = fmaxf(l0, fmaxf(l1, l2));
    float e0 = __expf(l0 - m), e1 = __expf(l1 - m), e2 = __expf(l2 - m);
    float s = e0 + e1 + e2;
    out[0] = e0 / s; out[1] = e1 / s; out[2] = e2 / s;
  }
}

// ---------------- host-side orchestration ----------------
static inline size_t alignup(size_t x) { return (x + 255) & ~(size_t)255; }

extern "C" void kernel_launch(void* const* d_in, const int* in_sizes, int n_in,
                              void* d_out, int out_size, void* d_ws, size_t ws_size,
                              hipStream_t stream) {
  (void)in_sizes; (void)n_in; (void)out_size; (void)ws_size;
  const float* x    = (const float*)d_in[0];
  const int*   ei   = (const int*)d_in[1];     // [2, E]
  const float* ea   = (const float*)d_in[2];
  const float* W1   = (const float*)d_in[3];
  const float* W2   = (const float*)d_in[4];
  const float* bias = (const float*)d_in[5];
  const float* Wih[3]  = { (const float*)d_in[6], (const float*)d_in[8],  (const float*)d_in[10] };
  const float* Whh[3]  = { (const float*)d_in[7], (const float*)d_in[9],  (const float*)d_in[11] };
  const float* linW = (const float*)d_in[12];
  const float* linb = (const float*)d_in[13];
  float* out = (float*)d_out;

  const int* erow = ei;
  const int* ecol = ei + EE;

  // ---- workspace carve-up (lifetime-aliased) ----
  char* p = (char*)d_ws;
  auto take = [&](size_t bytes) { char* r = p; p += alignup(bytes); return (void*)r; };
  float* xw      = (float*)take((size_t)NN * FOUT * 4);          //  8 MB
  float* agg     = (float*)take((size_t)NN * FOUT * 4);          //  8 MB
  float* deg     = (float*)take((size_t)NN * 4);
  float* dis     = (float*)take((size_t)NN * 4);
  float* w2sum   = (float*)take((size_t)FOUT * 4);
  unsigned short* W1b  = (unsigned short*)take((size_t)FOUT * FIN * 2);
  unsigned short* W2b  = (unsigned short*)take((size_t)FOUT * EDIM * 2);
  unsigned short* wihb = (unsigned short*)take((size_t)FOURH * HID * 2);   // reused x3
  unsigned short* seqb = (unsigned short*)take((size_t)NN * FOUT * 2);     //  4 MB
  float* h0s     = (float*)take((size_t)NN * HID * 4);           //  8 MB
  unsigned short* h0b = (unsigned short*)take((size_t)NN * HID * 2);
  float* h1s     = (float*)take((size_t)NN * HID * 4);           //  8 MB
  unsigned short* h1b = (unsigned short*)take((size_t)NN * HID * 2);
  float* raw     = (float*)take(256);
  // aliased: x_bf (8 MB, dead after xw GEMM)  <->  h2s (8 MB, written later)
  void* xbf_h2s  = take((size_t)NN * FIN * 2);                   //  8 MB
  unsigned short* xb  = (unsigned short*)xbf_h2s;
  float*          h2s = (float*)xbf_h2s;
  // aliased: ea_bf (32 MB, dead after edge kernel) <-> xp (32 MB)
  void* ea_xp    = take((size_t)EE * EDIM * 2);                  // 32 MB
  unsigned short* eab = (unsigned short*)ea_xp;
  float*          xp  = (float*)ea_xp;

  // ---- zero accumulators (buffers are NOT re-poisoned between replays) ----
  hipMemsetAsync(agg, 0, (size_t)NN * FOUT * 4, stream);
  hipMemsetAsync(raw, 0, 16, stream);

  // ---- degree / norm ----
  k_deg_init <<<(NN + 255) / 256, 256, 0, stream>>>(deg, NN);
  k_deg_accum<<<(EE + 255) / 256, 256, 0, stream>>>(ecol, deg, EE);
  k_dis      <<<(NN + 255) / 256, 256, 0, stream>>>(deg, dis, NN);

  // ---- bf16 conversions ----
  k_f32_to_bf16<<<2048, 256, 0, stream>>>(x,  xb,  (long)NN * FIN);
  k_f32_to_bf16<<<512,  256, 0, stream>>>(W1, W1b, (long)FOUT * FIN);
  k_f32_to_bf16<<<4096, 256, 0, stream>>>(ea, eab, (long)EE * EDIM);
  k_f32_to_bf16<<<128,  256, 0, stream>>>(W2, W2b, (long)FOUT * EDIM);
  k_w2_rowsum  <<<1, 256, 0, stream>>>(W2, w2sum);

  // ---- xw = x @ W1^T  (8192x512x256, WMMA + async-LDS weight staging) ----
  k_wmma_gemm_bf16<<<dim3(FOUT / 16, NN / 128), 256, 0, stream>>>(
      xb, W1b, xw, NN, FOUT, FIN);

  // ---- fused edge GEMM + message + scatter ----
  k_gcn_edges<<<EE / 16, 256, 0, stream>>>(eab, W2b, erow, ecol, dis, xw, agg);

  // ---- self loops + mean + bias + sigmoid -> bf16 sequence ----
  k_gcn_finalize<<<4096, 256, 0, stream>>>(agg, xw, w2sum, deg, bias, seqb);

  // ---- LSTM layer 0 ----
  k_f32_to_bf16<<<1024, 256, 0, stream>>>(Wih[0], wihb, (long)FOURH * HID);
  k_wmma_gemm_bf16<<<dim3(FOURH / 16, NN / 128), 256, 0, stream>>>(
      seqb, wihb, xp, NN, FOURH, HID);
  k_lstm_rec<<<1, 1024, 0, stream>>>(xp, Whh[0], h0s, h0b, NN);

  // ---- LSTM layer 1 ----
  k_f32_to_bf16<<<1024, 256, 0, stream>>>(Wih[1], wihb, (long)FOURH * HID);
  k_wmma_gemm_bf16<<<dim3(FOURH / 16, NN / 128), 256, 0, stream>>>(
      h0b, wihb, xp, NN, FOURH, HID);
  k_lstm_rec<<<1, 1024, 0, stream>>>(xp, Whh[1], h1s, h1b, NN);

  // ---- LSTM layer 2 ----
  k_f32_to_bf16<<<1024, 256, 0, stream>>>(Wih[2], wihb, (long)FOURH * HID);
  k_wmma_gemm_bf16<<<dim3(FOURH / 16, NN / 128), 256, 0, stream>>>(
      h1b, wihb, xp, NN, FOURH, HID);
  k_lstm_rec<<<1, 1024, 0, stream>>>(xp, Whh[2], h2s, (unsigned short*)nullptr, NN);

  // ---- pooling + linear + softmax ----
  k_lin_reduce<<<1024, 256, 0, stream>>>(h0s, h1s, h2s, linW, raw);
  k_softmax3<<<1, 32, 0, stream>>>(raw, linb, out);
}
